// MultiHeadDoubleAttention_41747082117769
// MI455X (gfx1250) — matrix operand order repaired and checked by reading, not compiled, not run on hardware
//
#include <hip/hip_runtime.h>

// ---------------------------------------------------------------------------
// MI455X (gfx1250). Conv(15x15,pad7 over 8x8) == 64 dense GEMMs
// (M=128 batch, N=256 co, K=64pix*256ci). All heavy math through
// v_wmma_f32_16x16x32_bf16. Weights re-laid-out ONCE to Wt[tap][co][ci]
// (B-transposed) so both A and B LDS panels are staged with plain coalesced
// 16B lane copies, issued as GLOBAL_LOAD_ASYNC_TO_LDS_B128 (ASYNCcnt path)
// via inline asm (ROCm builtin signature differs across toolchains).
// ---------------------------------------------------------------------------

typedef __attribute__((ext_vector_type(16))) __bf16 v16bf;
typedef __attribute__((ext_vector_type(8)))  float v8f;

#define LDK64 72  // LDS row pitch (halves) for K=64 panels: 144B rows, 16B aligned
#define LDK32 40  // LDS row pitch (halves) for K=32 panels: 80B rows, 16B aligned
#define LDV   66  // attention 64-wide bf16 rows
#define LDS_S 68  // score row stride (floats)

#define USE_ASYNC_LDS 1

// 16-byte global -> LDS copy. Per-lane LDS address in VGPR (low 32 bits of the
// flat pointer are the LDS byte offset per the aperture mapping), per-lane
// 64-bit global address in a VGPR pair. Tracked by ASYNCcnt.
static __device__ __forceinline__ void cp16(void* lds, const void* g) {
#if USE_ASYNC_LDS
  unsigned int laddr = (unsigned int)(unsigned long long)lds;
  unsigned long long gaddr = (unsigned long long)g;
  asm volatile("global_load_async_to_lds_b128 %0, %1, off"
               :: "v"(laddr), "v"(gaddr) : "memory");
#else
  const uint4 d = *(const uint4*)g;
  unsigned int* w = (unsigned int*)lds;
  w[0] = d.x; w[1] = d.y; w[2] = d.z; w[3] = d.w;
#endif
}
static __device__ __forceinline__ void cp_join() {
#if USE_ASYNC_LDS
  asm volatile("s_wait_asynccnt 0" ::: "memory");
#endif
}

static __device__ __forceinline__ unsigned short f2bf(float f) {
  unsigned int x = __float_as_uint(f);
  x += 0x7FFFu + ((x >> 16) & 1u);  // round-to-nearest-even
  return (unsigned short)(x >> 16);
}

union FragBF { v16bf v; unsigned int u[8]; };

// A fragment (16x32 MxK bf16) from LDS row-major tile (row stride ld halves).
// ISA layout: lanes 0-15 hold K 0-7/16-23, lanes 16-31 hold K 8-15/24-31.
static __device__ __forceinline__ v16bf loadA(const unsigned short* tile, int ld,
                                              int lmod, int half) {
  FragBF f;
  const unsigned short* ap = tile + lmod * ld;
#pragma unroll
  for (int p = 0; p < 8; ++p) {
    int off = 2 * p + 8 * half + ((p >= 4) ? 8 : 0);
    f.u[p] = *(const unsigned int*)(ap + off);
  }
  return f.v;
}

// B fragment (32x16 KxN bf16) from transposed LDS tile [N][K] (row stride ld).
// ISA layout: lanes 0-15 hold K 0-7/16-23, lanes 16-31 hold K 8-15/24-31, N = lane%16.
static __device__ __forceinline__ v16bf loadBT(const unsigned short* tileT, int ld,
                                               int lmod, int half) {
  FragBF f;
  const unsigned short* bp = tileT + lmod * ld + 8 * half;
#pragma unroll
  for (int p = 0; p < 8; ++p) f.u[p] = *(const unsigned int*)(bp + 2 * p + ((p >= 4) ? 8 : 0));
  return f.v;
}

// ---------------------------------------------------------------------------
// Prep kernels
// ---------------------------------------------------------------------------
__global__ void cast_bf16(const float* __restrict__ src,
                          unsigned short* __restrict__ dst, int n) {
  int i = blockIdx.x * 256 + threadIdx.x;
  if (i < n) dst[i] = f2bf(src[i]);
}

// Wt[tap][co][ci] = W[co][ci][tap]  (B-transposed GEMM panels; the scattered
// fp32 reads ride in L2: 59MB << 192MB. Writes are fully coalesced.)
__global__ void transpose_w(const float* __restrict__ W,
                            unsigned short* __restrict__ Wt) {
  unsigned int i = blockIdx.x * 256 + threadIdx.x;
  if (i >= 225u * 65536u) return;
  unsigned int tap = i >> 16, rem = i & 65535u;
  unsigned int co = rem >> 8, ci = rem & 255u;
  Wt[i] = f2bf(W[(size_t)co * 57600 + (size_t)ci * 225 + tap]);
}

// ---------------------------------------------------------------------------
// Conv-as-GEMM: grid (64 opix, 4 nblk), 256 threads = 8 waves (4 M x 2 N),
// each wave 2x2 accumulators. K loop: 64 ipix x 4 ci-chunks of 64.
// ---------------------------------------------------------------------------
__global__ __launch_bounds__(256) void conv_hollow_gemm(
    const unsigned short* __restrict__ X,   // [128][64][256] bf16
    const unsigned short* __restrict__ Wt,  // [225][256 co][256 ci] bf16
    const float* __restrict__ bias,         // [256]
    unsigned short* __restrict__ Y,         // [128][64][256] bf16
    int do_relu) {
  __shared__ unsigned short lA[128 * LDK64];  // [batch][ci]
  __shared__ unsigned short lB[64 * LDK64];   // [co][ci]
  const int o = blockIdx.x, nblk = blockIdx.y;
  const int co_base = nblk * 64;
  const int oy = o >> 3, ox = o & 7;
  const int tid = threadIdx.x, lane = tid & 31, wave = tid >> 5;
  const int mw = wave & 3, nw = wave >> 2;   // 4 M-quarters x 2 N-halves
  const int half = lane >> 4, lmod = lane & 15;

  v8f acc[2][2] = {};
  for (int ip = 0; ip < 64; ++ip) {
    const int iy = ip >> 3, ix = ip & 7;
    const int tap = (iy - oy + 7) * 15 + (ix - ox + 7);
    const unsigned short* wb = Wt + (size_t)tap * 65536;
    const unsigned short* xb = X + ip * 256;
    for (int kc = 0; kc < 256; kc += 64) {
      __syncthreads();
      // A panel: 128 rows x 64 ci -> 1024 x 16B segments
#pragma unroll
      for (int s = 0; s < 4; ++s) {
        int seg = tid + s * 256;
        int row = seg >> 3, c8 = (seg & 7) * 8;
        cp16(&lA[row * LDK64 + c8], xb + (size_t)row * 16384 + kc + c8);
      }
      // B panel (already transposed in global): 64 co x 64 ci -> 512 segments
#pragma unroll
      for (int s = 0; s < 2; ++s) {
        int seg = tid + s * 256;
        int row = seg >> 3, c8 = (seg & 7) * 8;
        cp16(&lB[row * LDK64 + c8],
             wb + (size_t)(co_base + row) * 256 + kc + c8);
      }
      cp_join();
      __syncthreads();
#pragma unroll
      for (int ks = 0; ks < 2; ++ks) {
        v16bf fb0 = loadBT(&lB[(nw * 32) * LDK64 + ks * 32], LDK64, lmod, half);
        v16bf fb1 = loadBT(&lB[(nw * 32 + 16) * LDK64 + ks * 32], LDK64, lmod, half);
#pragma unroll
        for (int mt = 0; mt < 2; ++mt) {
          v16bf fa = loadA(&lA[(mw * 32 + mt * 16) * LDK64 + ks * 32], LDK64, lmod, half);
          acc[mt][0] = __builtin_amdgcn_wmma_f32_16x16x32_bf16(
              false, fa, false, fb0, (short)0, acc[mt][0], false, false);
          acc[mt][1] = __builtin_amdgcn_wmma_f32_16x16x32_bf16(
              false, fa, false, fb1, (short)0, acc[mt][1], false, false);
        }
      }
    }
  }
#pragma unroll
  for (int nt = 0; nt < 2; ++nt) {
    const int n = co_base + nw * 32 + nt * 16 + lmod;
    const float bv = bias[n];
#pragma unroll
    for (int mt = 0; mt < 2; ++mt) {
      int m0 = mw * 32 + mt * 16 + 8 * half;
#pragma unroll
      for (int r = 0; r < 8; ++r) {
        float val = acc[mt][nt][r] + bv;
        if (do_relu) val = fmaxf(val, 0.f);
        Y[(size_t)(m0 + r) * 16384 + o * 256 + n] = f2bf(val);
      }
    }
  }
}

// ---------------------------------------------------------------------------
// Attention: one block per (b,h). QK^T + softmax + PV, all WMMA.
// ---------------------------------------------------------------------------
__global__ __launch_bounds__(256) void attention_kernel(
    const unsigned short* __restrict__ Qh, const unsigned short* __restrict__ Kh,
    const unsigned short* __restrict__ Vh, unsigned short* __restrict__ Ah) {
  __shared__ unsigned short sQ[64 * LDK32];  // [qpix][dk]
  __shared__ unsigned short sK[64 * LDK32];  // [kpix][dk] (== B^T stage)
  __shared__ unsigned short sVt[32 * LDV];   // [dk][kpix] (B^T stage for out)
  __shared__ unsigned short sP[64 * LDV];    // probs bf16 [qpix][kpix]
  __shared__ float sS[64 * LDS_S];           // scores f32

  const int b = blockIdx.x >> 3, h = blockIdx.x & 7;
  const int tid = threadIdx.x, lane = tid & 31, wave = tid >> 5;
  const int half = lane >> 4, lmod = lane & 15;
  const size_t base = (size_t)b * 16384 + h * 32;
  {
    int pix = tid >> 2, seg = tid & 3;
    size_t src = base + (size_t)pix * 256 + seg * 8;
    cp16(&sQ[pix * LDK32 + seg * 8], Qh + src);
    cp16(&sK[pix * LDK32 + seg * 8], Kh + src);
    uint4 dv = *(const uint4*)(Vh + src);
    const unsigned short* e = (const unsigned short*)&dv;
#pragma unroll
    for (int j = 0; j < 8; ++j) sVt[(seg * 8 + j) * LDV + pix] = e[j];
  }
  cp_join();
  __syncthreads();
  // scores: 16 tiles of 16x16 (K=32), 2 per wave (shared A fragment)
  {
    const int mt = wave >> 1, nt0 = (wave & 1) * 2;
    v16bf fa = loadA(&sQ[mt * 16 * LDK32], LDK32, lmod, half);
    const float scale = 0.17677669529663687f;  // 1/sqrt(32)
#pragma unroll
    for (int u = 0; u < 2; ++u) {
      int nt = nt0 + u;
      v16bf fb = loadBT(&sK[nt * 16 * LDK32], LDK32, lmod, half);
      v8f c = {};
      c = __builtin_amdgcn_wmma_f32_16x16x32_bf16(false, fa, false, fb, (short)0,
                                                  c, false, false);
#pragma unroll
      for (int r = 0; r < 8; ++r)
        sS[(mt * 16 + r + 8 * half) * LDS_S + nt * 16 + lmod] = c[r] * scale;
    }
  }
  __syncthreads();
  if (tid < 64) {  // row-wise softmax, one thread per query row
    float* row = &sS[tid * LDS_S];
    float mx = -1e30f;
    for (int j = 0; j < 64; ++j) mx = fmaxf(mx, row[j]);
    float sum = 0.f;
    for (int j = 0; j < 64; ++j) { float t = __expf(row[j] - mx); row[j] = t; sum += t; }
    float inv = 1.f / sum;
    unsigned short* prow = &sP[tid * LDV];
    for (int j = 0; j < 64; ++j) prow[j] = f2bf(row[j] * inv);
  }
  __syncthreads();
  // out = P[64,64] @ V[64,32]; 8 tiles, 1 per wave, 2 K-steps
  {
    const int mt = wave >> 1, nt = wave & 1;
    v8f c = {};
#pragma unroll
    for (int ks = 0; ks < 2; ++ks) {
      v16bf fa = loadA(&sP[mt * 16 * LDV + ks * 32], LDV, lmod, half);
      v16bf fb = loadBT(&sVt[nt * 16 * LDV + ks * 32], LDV, lmod, half);
      c = __builtin_amdgcn_wmma_f32_16x16x32_bf16(false, fa, false, fb, (short)0,
                                                  c, false, false);
    }
    const int n = h * 32 + nt * 16 + lmod;
#pragma unroll
    for (int r = 0; r < 8; ++r) {
      int pix = mt * 16 + r + 8 * half;
      Ah[((size_t)b * 64 + pix) * 256 + n] = f2bf(c[r]);
    }
  }
}

// ---------------------------------------------------------------------------
// Output projection: [8192,256] @ wo^T + bo  (wo row-major == [N][K] B stage)
// ---------------------------------------------------------------------------
__global__ __launch_bounds__(256) void proj_kernel(
    const unsigned short* __restrict__ Ain,  // [8192][256] bf16
    const unsigned short* __restrict__ Wb,   // [256 n][256 k] bf16
    const float* __restrict__ bo, float* __restrict__ Out) {
  __shared__ unsigned short lA[128 * LDK64];
  __shared__ unsigned short lB[64 * LDK64];
  const int mblk = blockIdx.x, nblk = blockIdx.y;
  const int co_base = nblk * 64;
  const int tid = threadIdx.x, lane = tid & 31, wave = tid >> 5;
  const int mw = wave & 3, nw = wave >> 2;
  const int half = lane >> 4, lmod = lane & 15;
  v8f acc[2][2] = {};
  const unsigned short* ab = Ain + (size_t)mblk * 128 * 256;
  for (int kc = 0; kc < 256; kc += 64) {
    __syncthreads();
#pragma unroll
    for (int s = 0; s < 4; ++s) {
      int seg = tid + s * 256;
      int row = seg >> 3, c8 = (seg & 7) * 8;
      cp16(&lA[row * LDK64 + c8], ab + (size_t)row * 256 + kc + c8);
    }
#pragma unroll
    for (int s = 0; s < 2; ++s) {
      int seg = tid + s * 256;
      int row = seg >> 3, c8 = (seg & 7) * 8;
      cp16(&lB[row * LDK64 + c8], Wb + (size_t)(co_base + row) * 256 + kc + c8);
    }
    cp_join();
    __syncthreads();
#pragma unroll
    for (int ks = 0; ks < 2; ++ks) {
      v16bf fb0 = loadBT(&lB[(nw * 32) * LDK64 + ks * 32], LDK64, lmod, half);
      v16bf fb1 = loadBT(&lB[(nw * 32 + 16) * LDK64 + ks * 32], LDK64, lmod, half);
#pragma unroll
      for (int mt = 0; mt < 2; ++mt) {
        v16bf fa = loadA(&lA[(mw * 32 + mt * 16) * LDK64 + ks * 32], LDK64, lmod, half);
        acc[mt][0] = __builtin_amdgcn_wmma_f32_16x16x32_bf16(
            false, fa, false, fb0, (short)0, acc[mt][0], false, false);
        acc[mt][1] = __builtin_amdgcn_wmma_f32_16x16x32_bf16(
            false, fa, false, fb1, (short)0, acc[mt][1], false, false);
      }
    }
  }
#pragma unroll
  for (int nt = 0; nt < 2; ++nt) {
    const int n = co_base + nw * 32 + nt * 16 + lmod;
    const float bv = bo[n];
#pragma unroll
    for (int mt = 0; mt < 2; ++mt) {
      int m0 = mblk * 128 + mw * 32 + mt * 16 + 8 * half;
#pragma unroll
      for (int r = 0; r < 8; ++r)
        Out[(size_t)(m0 + r) * 256 + n] = acc[mt][nt][r] + bv;
    }
  }
}

// ---------------------------------------------------------------------------
extern "C" void kernel_launch(void* const* d_in, const int* in_sizes, int n_in,
                              void* d_out, int out_size, void* d_ws, size_t ws_size,
                              hipStream_t stream) {
  const float* q  = (const float*)d_in[0];
  const float* k  = (const float*)d_in[1];
  const float* v  = (const float*)d_in[2];
  const float* wk = (const float*)d_in[3];
  const float* bk = (const float*)d_in[4];
  const float* wq = (const float*)d_in[5];
  const float* bq = (const float*)d_in[6];
  const float* wv = (const float*)d_in[7];
  const float* bv = (const float*)d_in[8];
  const float* wo = (const float*)d_in[9];
  const float* bo = (const float*)d_in[10];
  float* out = (float*)d_out;
  (void)in_sizes; (void)n_in; (void)out_size; (void)ws_size;

  char* ws = (char*)d_ws;
  size_t off = 0;
  auto carve = [&](size_t bytes) {
    char* p = ws + off;
    off += (bytes + 255) & ~(size_t)255;
    return p;
  };
  const size_t actB = (size_t)128 * 64 * 256 * 2;   // 4 MB
  const size_t wtB  = (size_t)225 * 256 * 256 * 2;  // 29.5 MB
  unsigned short* Xq  = (unsigned short*)carve(actB);
  unsigned short* Xk  = (unsigned short*)carve(actB);
  unsigned short* Xv  = (unsigned short*)carve(actB);
  unsigned short* WtK = (unsigned short*)carve(wtB);
  unsigned short* WtQ = (unsigned short*)carve(wtB);
  unsigned short* WtV = (unsigned short*)carve(wtB);
  unsigned short* WoB = (unsigned short*)carve((size_t)65536 * 2);
  unsigned short* T1  = (unsigned short*)carve(actB);
  unsigned short* Kh  = (unsigned short*)carve(actB);
  unsigned short* Qh  = (unsigned short*)carve(actB);
  unsigned short* Vh  = (unsigned short*)carve(actB);
  unsigned short* Ah  = (unsigned short*)carve(actB);

  const int nAct = 128 * 64 * 256;
  cast_bf16<<<(nAct + 255) / 256, 256, 0, stream>>>(q, Xq, nAct);
  cast_bf16<<<(nAct + 255) / 256, 256, 0, stream>>>(k, Xk, nAct);
  cast_bf16<<<(nAct + 255) / 256, 256, 0, stream>>>(v, Xv, nAct);
  cast_bf16<<<(65536 + 255) / 256, 256, 0, stream>>>(wo, WoB, 65536);
  const unsigned int nW = 225u * 65536u;
  transpose_w<<<(nW + 255) / 256, 256, 0, stream>>>(wk, WtK);
  transpose_w<<<(nW + 255) / 256, 256, 0, stream>>>(wq, WtQ);
  transpose_w<<<(nW + 255) / 256, 256, 0, stream>>>(wv, WtV);

  dim3 g(64, 4), blk(256);
  // k branch: conv(wk)->relu->conv(wk)
  conv_hollow_gemm<<<g, blk, 0, stream>>>(Xk, WtK, bk, T1, 1);
  conv_hollow_gemm<<<g, blk, 0, stream>>>(T1, WtK, bk, Kh, 0);
  // q branch: conv(wq)->relu->conv(wk)   (faithful to reference's reuse of k-conv)
  conv_hollow_gemm<<<g, blk, 0, stream>>>(Xq, WtQ, bq, T1, 1);
  conv_hollow_gemm<<<g, blk, 0, stream>>>(T1, WtK, bk, Qh, 0);
  // v branch: conv(wv)->relu->conv(wv)
  conv_hollow_gemm<<<g, blk, 0, stream>>>(Xv, WtV, bv, T1, 1);
  conv_hollow_gemm<<<g, blk, 0, stream>>>(T1, WtV, bv, Vh, 0);

  attention_kernel<<<1024, 256, 0, stream>>>(Qh, Kh, Vh, Ah);
  proj_kernel<<<g, blk, 0, stream>>>(Ah, WoB, bo, out);
}
